// ResidualCache_54958401520013
// MI455X (gfx1250) — matrix operand order repaired and checked by reading, not compiled, not run on hardware
//
#include <hip/hip_runtime.h>
#include <hip/hip_bf16.h>
#include <math.h>

typedef __attribute__((ext_vector_type(16))) _Float16 v16h;
typedef __attribute__((ext_vector_type(8)))  _Float16 v8h;
typedef __attribute__((ext_vector_type(8)))  float    v8f;

union H16 { v16h v; struct { v8h lo, hi; } s; };
union HP4 { _Float16 h[4]; uint2 u; };

#define DD 1024
#define NEG_INF (-1e30f)

__device__ __forceinline__ void topk_insert(float s, int idx,
    float& s0, float& s1, float& s2, float& s3,
    int& i0, int& i1, int& i2, int& i3) {
  if (s > s3) {
    s3 = s; i3 = idx;
    if (s3 > s2) { float tf = s2; s2 = s3; s3 = tf; int ti = i2; i2 = i3; i3 = ti; }
    if (s2 > s1) { float tf = s1; s1 = s2; s2 = tf; int ti = i1; i1 = i2; i2 = ti; }
    if (s1 > s0) { float tf = s0; s0 = s1; s1 = tf; int ti = i0; i0 = i1; i1 = ti; }
  }
}

// ---------------- Kernel 1: normalize keys -> f16 ----------------
__global__ __launch_bounds__(256) void rc_knorm(const float* __restrict__ K,
                                                _Float16* __restrict__ Kh) {
  int row  = blockIdx.x * 8 + (threadIdx.x >> 5);
  int lane = threadIdx.x & 31;
  const float4* rp = (const float4*)(K + (size_t)row * DD);
  float4 rg[8];
  float ss = 0.0f;
#pragma unroll
  for (int i = 0; i < 8; i++) {
    float4 t = rp[lane + 32 * i];
    rg[i] = t;
    ss += t.x * t.x + t.y * t.y + t.z * t.z + t.w * t.w;
  }
#pragma unroll
  for (int o = 16; o > 0; o >>= 1) ss += __shfl_xor(ss, o, 32);
  float scale = 1.0f / fmaxf(sqrtf(ss), 1e-12f);
  uint2* op = (uint2*)(Kh + (size_t)row * DD);
#pragma unroll
  for (int i = 0; i < 8; i++) {
    HP4 p;
    p.h[0] = (_Float16)(rg[i].x * scale);
    p.h[1] = (_Float16)(rg[i].y * scale);
    p.h[2] = (_Float16)(rg[i].z * scale);
    p.h[3] = (_Float16)(rg[i].w * scale);
    op[lane + 32 * i] = p.u;
  }
}

// ---------------- Kernel 2: fused sim GEMM (WMMA f16) + partial top-4 ----------------
// grid = (B/64)*2 blocks, 256 threads (8 waves). Block (qb, sp): rows qb*64..+63,
// keys [sp*4096, sp*4096+4096).
// Wave w: mg = w>>2 (M-group: 2 tiles, rows mg*32..+31), nj = w&3 (N phase).
// Per wave per chunk: 2 A-tiles x 4 B-tiles -> 8 WMMAs per K-step from
// 4 ds_load_b128 + 8 global_load_b128 (1.5 mem ops / WMMA).
__global__ __launch_bounds__(256) void rc_simtopk(const float* __restrict__ Q,
                                                  const _Float16* __restrict__ Kh,
                                                  float* __restrict__ cand_s,
                                                  int* __restrict__ cand_i) {
  __shared__ _Float16 qt[64][1032];        // padded stride: conflict-free A-frag reads
  __shared__ float simbuf[8][8][16][20];   // per-wave sim-tile transpose scratch

  const int qb = blockIdx.x >> 1;
  const int sp = blockIdx.x & 1;
  const int m0 = qb * 64;
  const int w    = threadIdx.x >> 5;
  const int lane = threadIdx.x & 31;

  // ---- normalize 64 Q rows into LDS (f16), 8 rows per wave ----
#pragma unroll 1
  for (int rr = 0; rr < 8; rr++) {
    int lr = w * 8 + rr;
    const float4* rp = (const float4*)(Q + (size_t)(m0 + lr) * DD);
    float4 rg[8];
    float ss = 0.0f;
#pragma unroll
    for (int i = 0; i < 8; i++) {
      float4 t = rp[lane + 32 * i];
      rg[i] = t;
      ss += t.x * t.x + t.y * t.y + t.z * t.z + t.w * t.w;
    }
#pragma unroll
    for (int o = 16; o > 0; o >>= 1) ss += __shfl_xor(ss, o, 32);
    float scale = 1.0f / fmaxf(sqrtf(ss), 1e-12f);
#pragma unroll
    for (int i = 0; i < 8; i++) {
      HP4 p;
      p.h[0] = (_Float16)(rg[i].x * scale);
      p.h[1] = (_Float16)(rg[i].y * scale);
      p.h[2] = (_Float16)(rg[i].z * scale);
      p.h[3] = (_Float16)(rg[i].w * scale);
      *(uint2*)&qt[lr][4 * (lane + 32 * i)] = p.u;
    }
  }
  __syncthreads();

  const int mg = w >> 2;                 // M-group: rows mg*32 .. mg*32+31 (2 tiles)
  const int nj = w & 3;                  // N phase
  const int h  = lane >> 4;              // lane half
  const int lr15 = lane & 15;
  const int arow0 = mg * 32 + lr15;      // A-tile 0 row in LDS
  const int arow1 = arow0 + 16;          // A-tile 1 row in LDS
  const int aoff = (lane < 16) ? 0 : 8;  // 16-bit A layout: lanes>=16 hold K+8
  const int boff = (lane < 16) ? 0 : 8;  // 16-bit B layout: lanes>=16 hold K+8

  // two independent top-4 sets (one per owned row in each A-tile)
  float as0 = NEG_INF, as1 = NEG_INF, as2 = NEG_INF, as3 = NEG_INF;
  int   ai0 = 0, ai1 = 0, ai2 = 0, ai3 = 0;
  float bs0 = NEG_INF, bs1 = NEG_INF, bs2 = NEG_INF, bs3 = NEG_INF;
  int   bi0 = 0, bi1 = 0, bi2 = 0, bi3 = 0;

#pragma unroll 1
  for (int t = 0; t < 16; t++) {
    const int c = nj + 4 * t;                  // 64-key chunk index (64 chunks/split)
    const int nbase = sp * 4096 + c * 64;
    v8f acc0[4] = {};
    v8f acc1[4] = {};
    const _Float16* bbase = Kh + (size_t)(nbase + lr15) * DD + boff;

#pragma unroll 2
    for (int kb = 0; kb < DD; kb += 32) {
      H16 a0, a1;
      a0.s.lo = *(const v8h*)&qt[arow0][kb + aoff];
      a0.s.hi = *(const v8h*)&qt[arow0][kb + 16 + aoff];
      a1.s.lo = *(const v8h*)&qt[arow1][kb + aoff];
      a1.s.hi = *(const v8h*)&qt[arow1][kb + 16 + aoff];
#pragma unroll
      for (int tl = 0; tl < 4; tl++) {
        const _Float16* bp = bbase + (size_t)tl * 16 * DD + kb;
        H16 b;
        b.s.lo = *(const v8h*)bp;
        b.s.hi = *(const v8h*)(bp + 16);
        acc0[tl] = __builtin_amdgcn_wmma_f32_16x16x32_f16(
            false, a0.v, false, b.v, (short)0, acc0[tl], false, false);
        acc1[tl] = __builtin_amdgcn_wmma_f32_16x16x32_f16(
            false, a1.v, false, b.v, (short)0, acc1[tl], false, false);
      }
    }

    // transpose sim tiles through LDS: element (m = j+8h, n = lane&15)
#pragma unroll
    for (int tl = 0; tl < 4; tl++)
#pragma unroll
      for (int j = 0; j < 8; j++) {
        simbuf[w][tl][j + 8 * h][lr15]     = acc0[tl][j];
        simbuf[w][4 + tl][j + 8 * h][lr15] = acc1[tl][j];
      }
    asm volatile("s_wait_dscnt 0" ::: "memory");

    // each lane scans 2 N-tiles (32 keys) for its own row in EACH A-tile
#pragma unroll
    for (int tt = 0; tt < 2; tt++) {
      const int tl = 2 * h + tt;
      const int kbase = nbase + tl * 16;
      const float4* sr0 = (const float4*)&simbuf[w][tl][lr15][0];
      const float4* sr1 = (const float4*)&simbuf[w][4 + tl][lr15][0];
#pragma unroll
      for (int qq = 0; qq < 4; qq++) {
        float4 sv = sr0[qq];
        int base = kbase + qq * 4;
        topk_insert(sv.x, base + 0, as0, as1, as2, as3, ai0, ai1, ai2, ai3);
        topk_insert(sv.y, base + 1, as0, as1, as2, as3, ai0, ai1, ai2, ai3);
        topk_insert(sv.z, base + 2, as0, as1, as2, as3, ai0, ai1, ai2, ai3);
        topk_insert(sv.w, base + 3, as0, as1, as2, as3, ai0, ai1, ai2, ai3);
      }
#pragma unroll
      for (int qq = 0; qq < 4; qq++) {
        float4 sv = sr1[qq];
        int base = kbase + qq * 4;
        topk_insert(sv.x, base + 0, bs0, bs1, bs2, bs3, bi0, bi1, bi2, bi3);
        topk_insert(sv.y, base + 1, bs0, bs1, bs2, bs3, bi0, bi1, bi2, bi3);
        topk_insert(sv.z, base + 2, bs0, bs1, bs2, bs3, bi0, bi1, bi2, bi3);
        topk_insert(sv.w, base + 3, bs0, bs1, bs2, bs3, bi0, bi1, bi2, bi3);
      }
    }
    asm volatile("s_wait_dscnt 0" ::: "memory");
  }

  // write partial candidates: 16 sets of 4 per row across (sp, nj, half)
  const int set = sp * 8 + nj * 2 + h;
  {
    const int row_g = m0 + mg * 32 + lr15;
    size_t o = (size_t)row_g * 64 + set * 4;
    cand_s[o + 0] = as0; cand_s[o + 1] = as1; cand_s[o + 2] = as2; cand_s[o + 3] = as3;
    cand_i[o + 0] = ai0; cand_i[o + 1] = ai1; cand_i[o + 2] = ai2; cand_i[o + 3] = ai3;
  }
  {
    const int row_g = m0 + mg * 32 + 16 + lr15;
    size_t o = (size_t)row_g * 64 + set * 4;
    cand_s[o + 0] = bs0; cand_s[o + 1] = bs1; cand_s[o + 2] = bs2; cand_s[o + 3] = bs3;
    cand_i[o + 0] = bi0; cand_i[o + 1] = bi1; cand_i[o + 2] = bi2; cand_i[o + 3] = bi3;
  }
}

// ---------------- Kernel 3: merge -> softmax -> gather V ----------------
__global__ __launch_bounds__(256) void rc_gather(const float* __restrict__ cand_s,
                                                 const int* __restrict__ cand_i,
                                                 const float* __restrict__ V,
                                                 float* __restrict__ out) {
  const int b = blockIdx.x;
  __shared__ float w4[4];
  __shared__ int   id4[4];
  if (threadIdx.x == 0) {
    float s0 = NEG_INF, s1 = NEG_INF, s2 = NEG_INF, s3 = NEG_INF;
    int i0 = 0, i1 = 0, i2 = 0, i3 = 0;
    const float* cs = cand_s + (size_t)b * 64;
    const int*   ci = cand_i + (size_t)b * 64;
    for (int j = 0; j < 64; j++)
      topk_insert(cs[j], ci[j], s0, s1, s2, s3, i0, i1, i2, i3);
    float m = s0;  // sorted descending
    float e0 = expf(s0 - m), e1 = expf(s1 - m), e2 = expf(s2 - m), e3 = expf(s3 - m);
    float inv = 1.0f / (e0 + e1 + e2 + e3);
    w4[0] = e0 * inv; w4[1] = e1 * inv; w4[2] = e2 * inv; w4[3] = e3 * inv;
    id4[0] = i0; id4[1] = i1; id4[2] = i2; id4[3] = i3;
  }
  __syncthreads();
  float w0 = w4[0], w1 = w4[1], w2 = w4[2], w3 = w4[3];
  const float4* v0 = (const float4*)(V + (size_t)id4[0] * DD);
  const float4* v1 = (const float4*)(V + (size_t)id4[1] * DD);
  const float4* v2 = (const float4*)(V + (size_t)id4[2] * DD);
  const float4* v3 = (const float4*)(V + (size_t)id4[3] * DD);
  float4* o = (float4*)(out + (size_t)b * DD);
  for (int cidx = threadIdx.x; cidx < DD / 4; cidx += blockDim.x) {
    float4 x0 = v0[cidx], x1 = v1[cidx], x2 = v2[cidx], x3 = v3[cidx];
    float4 r;
    r.x = w0 * x0.x + w1 * x1.x + w2 * x2.x + w3 * x3.x;
    r.y = w0 * x0.y + w1 * x1.y + w2 * x2.y + w3 * x3.y;
    r.z = w0 * x0.z + w1 * x1.z + w2 * x2.z + w3 * x3.z;
    r.w = w0 * x0.w + w1 * x1.w + w2 * x2.w + w3 * x3.w;
    o[cidx] = r;
  }
}

extern "C" void kernel_launch(void* const* d_in, const int* in_sizes, int n_in,
                              void* d_out, int out_size, void* d_ws, size_t ws_size,
                              hipStream_t stream) {
  const float* Q = (const float*)d_in[0];
  const float* K = (const float*)d_in[1];
  const float* V = (const float*)d_in[2];
  float* out = (float*)d_out;
  const int B = in_sizes[0] / DD;   // 16384
  const int N = in_sizes[1] / DD;   // 8192

  _Float16* Kh  = (_Float16*)d_ws;                               // N*D*2  = 16 MB
  float* cand_s = (float*)((char*)d_ws + (size_t)N * DD * 2);    // B*64*4 =  4 MB
  int*   cand_i = (int*)(cand_s + (size_t)B * 64);               // B*64*4 =  4 MB

  rc_knorm<<<N / 8, 256, 0, stream>>>(K, Kh);
  rc_simtopk<<<(B / 64) * 2, 256, 0, stream>>>(Q, Kh, cand_s, cand_i);
  rc_gather<<<B, 256, 0, stream>>>(cand_s, cand_i, V, out);
}
